// MultiHeadAttention_68212670595145
// MI455X (gfx1250) — matrix operand order repaired and checked by reading, not compiled, not run on hardware
//
#include <hip/hip_runtime.h>
#include <hip/hip_bf16.h>
#include <math.h>

#define BGLOB 2
#define SLEN  2048
#define DM    1024
#define NH    16
#define DH    64
#define NTOK  (BGLOB * SLEN)        // 4096 tokens

typedef __attribute__((ext_vector_type(16))) __bf16 v16bf;
typedef __attribute__((ext_vector_type(8)))  __bf16 v8bf;
typedef __attribute__((ext_vector_type(8)))  float  v8f;
typedef __attribute__((ext_vector_type(4)))  float  v4f;

// ---------------------------------------------------------------------------
// WMMA operand builders.
// A layout (16x32 bf16): lane L: M = L&15, hi = L>>4.
//   elems 0..15 = K (16*hi + 0..15).
//   -> two contiguous 8-element runs: at p and p+8 (relative K-run start).
// B layout (32x16 bf16): lane L: N = L&15, hi = L>>4.
//   elems 0..15 = K (16*hi + 0..15)  -> one contiguous 16-element run.
// C/D layout (16x16 f32): VGPR r: M = r + 8*hi, N = L&15.
// ---------------------------------------------------------------------------

__device__ __forceinline__ v16bf make_a_from_f32(const float* p) {
  const v4f f0 = *(const v4f*)(p);
  const v4f f1 = *(const v4f*)(p + 4);
  const v4f f2 = *(const v4f*)(p + 8);
  const v4f f3 = *(const v4f*)(p + 12);
  v16bf a;
  a[0]=(__bf16)f0.x;  a[1]=(__bf16)f0.y;  a[2]=(__bf16)f0.z;  a[3]=(__bf16)f0.w;
  a[4]=(__bf16)f1.x;  a[5]=(__bf16)f1.y;  a[6]=(__bf16)f1.z;  a[7]=(__bf16)f1.w;
  a[8]=(__bf16)f2.x;  a[9]=(__bf16)f2.y;  a[10]=(__bf16)f2.z; a[11]=(__bf16)f2.w;
  a[12]=(__bf16)f3.x; a[13]=(__bf16)f3.y; a[14]=(__bf16)f3.z; a[15]=(__bf16)f3.w;
  return a;
}

__device__ __forceinline__ v16bf make_b_from_f32(const float* __restrict__ p) {
  const v4f f0 = *(const v4f*)(p);
  const v4f f1 = *(const v4f*)(p + 4);
  const v4f f2 = *(const v4f*)(p + 8);
  const v4f f3 = *(const v4f*)(p + 12);
  v16bf b;
  b[0]=(__bf16)f0.x;  b[1]=(__bf16)f0.y;  b[2]=(__bf16)f0.z;  b[3]=(__bf16)f0.w;
  b[4]=(__bf16)f1.x;  b[5]=(__bf16)f1.y;  b[6]=(__bf16)f1.z;  b[7]=(__bf16)f1.w;
  b[8]=(__bf16)f2.x;  b[9]=(__bf16)f2.y;  b[10]=(__bf16)f2.z; b[11]=(__bf16)f2.w;
  b[12]=(__bf16)f3.x; b[13]=(__bf16)f3.y; b[14]=(__bf16)f3.z; b[15]=(__bf16)f3.w;
  return b;
}

__device__ __forceinline__ v16bf make_a_from_bf16(const __bf16* __restrict__ p) {
  v8bf lo = *(const v8bf*)(p);
  v8bf hi = *(const v8bf*)(p + 8);
  return __builtin_shufflevector(lo, hi, 0,1,2,3,4,5,6,7,8,9,10,11,12,13,14,15);
}

__device__ __forceinline__ v8f wmma_bf16(v16bf a, v16bf b, v8f c) {
  return __builtin_amdgcn_wmma_f32_16x16x32_bf16(false, a, false, b, (short)0, c,
                                                 false, false);
}

// Async global->LDS copy (CDNA5). lds = byte offset in workgroup LDS, g =
// per-lane global address. Tracked by ASYNCcnt.
__device__ __forceinline__ void async_b128(unsigned lds, const void* g) {
  asm volatile("global_load_async_to_lds_b128 %0, %1, off"
               :: "v"(lds), "v"(g) : "memory");
}

// ---------------------------------------------------------------------------
// Kernel 1: fused QKV projection.  For each of q/k/v:  y = x @ W^T + b.
// One wave computes a 32(M) x 64(N) block.  q,k -> [B,H,S,DH] bf16 row-major;
// v -> [B,H,DH,S] bf16 (transposed) so attn@v has a K-contiguous B operand.
// ---------------------------------------------------------------------------
__global__ __launch_bounds__(256) void qkv_kernel(
    const float* __restrict__ xq, const float* __restrict__ xk,
    const float* __restrict__ xv,
    const float* __restrict__ Wq, const float* __restrict__ bq,
    const float* __restrict__ Wk, const float* __restrict__ bk,
    const float* __restrict__ Wv, const float* __restrict__ bv,
    __bf16* __restrict__ qbuf, __bf16* __restrict__ kbuf,
    __bf16* __restrict__ vTbuf) {
  const int wave = threadIdx.x >> 5;
  const int lane = threadIdx.x & 31;
  const int hi = lane >> 4, l15 = lane & 15;
  const int gw = blockIdx.x * 8 + wave;
  const int NB = (3 * DM) / 64;                 // 48 column blocks
  const int mb = gw / NB, nb = gw % NB;
  const int m0 = mb * 32;
  const int c0g = nb * 64;
  const int mi = c0g / DM;                      // 0=q, 1=k, 2=v
  const int c0 = c0g % DM;

  const float* X = (mi == 0) ? xq : (mi == 1) ? xk : xv;
  const float* W = (mi == 0) ? Wq : (mi == 1) ? Wk : Wv;
  const float* bias = (mi == 0) ? bq : (mi == 1) ? bk : bv;

  v8f acc[2][4];
#pragma unroll
  for (int i = 0; i < 2; ++i)
#pragma unroll
    for (int j = 0; j < 4; ++j) acc[i][j] = {};

  for (int k0 = 0; k0 < DM; k0 += 32) {
    v16bf A[2], Bm[4];
#pragma unroll
    for (int i = 0; i < 2; ++i)
      A[i] = make_a_from_f32(X + (size_t)(m0 + i * 16 + l15) * DM + k0 + 16 * hi);
#pragma unroll
    for (int j = 0; j < 4; ++j)
      Bm[j] = make_b_from_f32(W + (size_t)(c0 + j * 16 + l15) * DM + k0 + 16 * hi);
#pragma unroll
    for (int i = 0; i < 2; ++i)
#pragma unroll
      for (int j = 0; j < 4; ++j) acc[i][j] = wmma_bf16(A[i], Bm[j], acc[i][j]);
  }

  if (mi < 2) {
    __bf16* dst = (mi == 0) ? qbuf : kbuf;
#pragma unroll
    for (int i = 0; i < 2; ++i)
#pragma unroll
      for (int j = 0; j < 4; ++j) {
        const int col = c0 + j * 16 + l15;
        const int h = col >> 6, f = col & 63;
        const float bb = bias[col];
#pragma unroll
        for (int r = 0; r < 8; ++r) {
          const int tok = m0 + i * 16 + 8 * hi + r;
          const int bidx = tok >> 11, s = tok & (SLEN - 1);
          dst[(((size_t)bidx * NH + h) * SLEN + s) * DH + f] =
              (__bf16)(acc[i][j][r] + bb);
        }
      }
  } else {
#pragma unroll
    for (int i = 0; i < 2; ++i)
#pragma unroll
      for (int j = 0; j < 4; ++j) {
        const int col = c0 + j * 16 + l15;
        const int h = col >> 6, f = col & 63;
        const float bb = bias[col];
        const int t0 = m0 + i * 16 + 8 * hi;    // 8 consecutive tokens
        const int bidx = t0 >> 11, s0 = t0 & (SLEN - 1);
        v8bf pk;
#pragma unroll
        for (int r = 0; r < 8; ++r) pk[r] = (__bf16)(acc[i][j][r] + bb);
        *(v8bf*)(vTbuf + (((size_t)bidx * NH + h) * DH + f) * SLEN + s0) = pk;
      }
  }
}

// ---------------------------------------------------------------------------
// Kernel 2: attention for one (b,h, 16-query tile).  128 threads = 4 waves.
// LDS: 16 x 2048 f32 score block (128 KB) + row stats + 8 KB async vT stage.
// Phases:
//   A) WMMA scores -> LDS
//   B) row max, then e = exp(s-max) stored back to LDS, row sum -> rinv
//   C) attn = e * rinv, nontemporal float4 stores (537 MB stream, bypass L2)
//   D) ctx = (e @ vT) * rinv via WMMA; vT chunks double-buffered into LDS
//      with global_load_async_to_lds_b128 + s_wait_asynccnt.
// ---------------------------------------------------------------------------
__global__ __launch_bounds__(128) void attn_kernel(
    const __bf16* __restrict__ qbuf, const __bf16* __restrict__ kbuf,
    const __bf16* __restrict__ vTbuf, __bf16* __restrict__ ctxbuf,
    float* __restrict__ attn_out) {
  extern __shared__ float sc[];                 // [16][SLEN] + stats + stage
  float* rowrin = sc + 16 * SLEN;               // 16 floats (+16 pad)
  __bf16* vstage = (__bf16*)(rowrin + 32);      // 4 waves x 2 bufs x 512 bf16

  const int blk = blockIdx.x;                   // bh*128 + qb
  const int qb = blk & 127;
  const int bh = blk >> 7;                      // b*NH + h
  const int q0 = qb * 16;
  const int wave = threadIdx.x >> 5;
  const int lane = threadIdx.x & 31;
  const int hi = lane >> 4, l15 = lane & 15;

  const __bf16* qh = qbuf + (size_t)bh * SLEN * DH;
  const __bf16* kh = kbuf + (size_t)bh * SLEN * DH;
  const __bf16* vh = vTbuf + (size_t)bh * DH * SLEN;

  // ---- Phase A: scores ----------------------------------------------------
  v16bf qa[2];
#pragma unroll
  for (int t = 0; t < 2; ++t)
    qa[t] = make_a_from_bf16(qh + (size_t)(q0 + l15) * DH + t * 32 + 16 * hi);

  for (int nt = wave; nt < SLEN / 16; nt += 4) {
    const int n0 = nt * 16;
    const __bf16* kp = kh + (size_t)(n0 + l15) * DH;
    v16bf kb0 = *(const v16bf*)(kp + 16 * hi);        // d = 0..31
    v16bf kb1 = *(const v16bf*)(kp + 32 + 16 * hi);   // d = 32..63
    v8f s = {};
    s = wmma_bf16(qa[0], kb0, s);
    s = wmma_bf16(qa[1], kb1, s);
#pragma unroll
    for (int r = 0; r < 8; ++r)
      sc[(r + 8 * hi) * SLEN + n0 + l15] = s[r] * 0.125f;   // 1/sqrt(64)
  }
  __syncthreads();

  // ---- Phase B: row max, e = exp(s-max) in place, rinv --------------------
  {
    const int row = threadIdx.x >> 3;
    const int sub = threadIdx.x & 7;
    float* rp = sc + row * SLEN + sub * 256;
    float mx = -INFINITY;
    for (int k = 0; k < 256; k += 4) {
      const v4f v = *(const v4f*)(rp + k);
      mx = fmaxf(mx, fmaxf(fmaxf(v.x, v.y), fmaxf(v.z, v.w)));
    }
#pragma unroll
    for (int d = 1; d < 8; d <<= 1) mx = fmaxf(mx, __shfl_xor(mx, d));
    float sum = 0.f;
    for (int k = 0; k < 256; k += 4) {
      const v4f v = *(const v4f*)(rp + k);
      v4f e;
      e.x = __expf(v.x - mx); e.y = __expf(v.y - mx);
      e.z = __expf(v.z - mx); e.w = __expf(v.w - mx);
      sum += (e.x + e.y) + (e.z + e.w);
      *(v4f*)(rp + k) = e;                     // store unnormalized exp
    }
#pragma unroll
    for (int d = 1; d < 8; d <<= 1) sum += __shfl_xor(sum, d);
    if (sub == 0) rowrin[row] = 1.0f / sum;
  }
  __syncthreads();

  // ---- Phase C: attn = e * rinv, nontemporal stream to global -------------
  float* attn_blk = attn_out + ((size_t)bh * SLEN + q0) * SLEN;
  for (int i = threadIdx.x * 4; i < 16 * SLEN; i += 128 * 4) {
    const float r = rowrin[i >> 11];
    const v4f e = *(const v4f*)(sc + i);
    v4f p;
    p.x = e.x * r; p.y = e.y * r; p.z = e.z * r; p.w = e.w * r;
    __builtin_nontemporal_store(p, (v4f*)(attn_blk + i));
  }

  // ---- Phase D: ctx = (e @ vT) * rinv, async double-buffered vT -----------
  const int n = wave;                            // d tile 0..3
  __bf16* mybuf = vstage + wave * 1024;          // 2 bufs x 512 bf16 (1KB ea.)
  const unsigned ldsbase = (unsigned)(size_t)mybuf;
  const unsigned slot = (unsigned)(l15 * 64 + hi * 32);   // byte slot in buf
  const __bf16* gsrc = vh + (size_t)(n * 16 + l15) * SLEN + 16 * hi;

  async_b128(ldsbase + slot, gsrc);              // prologue: chunk kk=0
  async_b128(ldsbase + slot + 16, (const char*)gsrc + 16);

  v8f acc = {};
  for (int kk = 0; kk < SLEN; kk += 32) {
    const int cur = (kk >> 5) & 1;
    if (kk + 32 < SLEN) {                        // prefetch next chunk
      const __bf16* gn = gsrc + (kk + 32);
      const unsigned ln = ldsbase + (unsigned)((cur ^ 1) * 1024) + slot;
      async_b128(ln, gn);
      async_b128(ln + 16, (const char*)gn + 16);
      asm volatile("s_wait_asynccnt 0x2" ::: "memory");   // cur buf landed
    } else {
      asm volatile("s_wait_asynccnt 0x0" ::: "memory");
    }
    const __bf16* bp = mybuf + cur * 512 + l15 * 32 + 16 * hi;
    v16bf Bm = *(const v16bf*)bp;
    v16bf A = make_a_from_f32(sc + (size_t)l15 * SLEN + kk + 16 * hi);
    acc = wmma_bf16(A, Bm, acc);
  }

  const int b_ = bh >> 4, h_ = bh & 15;
#pragma unroll
  for (int r = 0; r < 8; ++r) {
    const float rs = rowrin[r + 8 * hi];         // fold softmax scale in here
    const int s = q0 + r + 8 * hi;
    const size_t tok = (size_t)b_ * SLEN + s;
    ctxbuf[tok * DM + h_ * DH + n * 16 + l15] = (__bf16)(acc[r] * rs);
  }
}

// ---------------------------------------------------------------------------
// Kernel 3: out = ctx @ Wo^T + bo  (f32 result into d_out, nontemporal).
// ---------------------------------------------------------------------------
__global__ __launch_bounds__(256) void out_kernel(
    const __bf16* __restrict__ ctxbuf, const float* __restrict__ Wo,
    const float* __restrict__ bo, float* __restrict__ out) {
  const int wave = threadIdx.x >> 5;
  const int lane = threadIdx.x & 31;
  const int hi = lane >> 4, l15 = lane & 15;
  const int gw = blockIdx.x * 8 + wave;
  const int NB = DM / 64;                       // 16 column blocks
  const int mb = gw / NB, nb = gw % NB;
  const int m0 = mb * 32, c0 = nb * 64;

  v8f acc[2][4];
#pragma unroll
  for (int i = 0; i < 2; ++i)
#pragma unroll
    for (int j = 0; j < 4; ++j) acc[i][j] = {};

  for (int k0 = 0; k0 < DM; k0 += 32) {
    v16bf A[2], Bm[4];
#pragma unroll
    for (int i = 0; i < 2; ++i)
      A[i] = make_a_from_bf16(ctxbuf + (size_t)(m0 + i * 16 + l15) * DM + k0 + 16 * hi);
#pragma unroll
    for (int j = 0; j < 4; ++j)
      Bm[j] = make_b_from_f32(Wo + (size_t)(c0 + j * 16 + l15) * DM + k0 + 16 * hi);
#pragma unroll
    for (int i = 0; i < 2; ++i)
#pragma unroll
      for (int j = 0; j < 4; ++j) acc[i][j] = wmma_bf16(A[i], Bm[j], acc[i][j]);
  }

#pragma unroll
  for (int i = 0; i < 2; ++i)
#pragma unroll
    for (int j = 0; j < 4; ++j) {
      const int col = c0 + j * 16 + l15;
      const float bb = bo[col];
#pragma unroll
      for (int r = 0; r < 8; ++r) {
        const int tok = m0 + i * 16 + 8 * hi + r;
        __builtin_nontemporal_store(acc[i][j][r] + bb,
                                    out + (size_t)tok * DM + col);
      }
    }
}

// ---------------------------------------------------------------------------
extern "C" void kernel_launch(void* const* d_in, const int* in_sizes, int n_in,
                              void* d_out, int out_size, void* d_ws,
                              size_t ws_size, hipStream_t stream) {
  (void)in_sizes; (void)n_in; (void)out_size; (void)ws_size;
  const float* xq = (const float*)d_in[0];
  const float* xk = (const float*)d_in[1];
  const float* xv = (const float*)d_in[2];
  const float* Wq = (const float*)d_in[3];
  const float* bq = (const float*)d_in[4];
  const float* Wk = (const float*)d_in[5];
  const float* bk = (const float*)d_in[6];
  const float* Wv = (const float*)d_in[7];
  const float* bv = (const float*)d_in[8];
  const float* Wo = (const float*)d_in[9];
  const float* bo = (const float*)d_in[10];

  const size_t QSZ = (size_t)BGLOB * NH * SLEN * DH;   // 4194304 elems
  __bf16* qbuf = (__bf16*)d_ws;
  __bf16* kbuf = qbuf + QSZ;
  __bf16* vTbuf = kbuf + QSZ;
  __bf16* ctxbuf = vTbuf + QSZ;

  float* out = (float*)d_out;
  float* attn = out + (size_t)BGLOB * SLEN * DM;       // attn after out

  // Kernel 1: (4096/32) * 48 = 6144 waves -> 768 blocks of 8 waves
  qkv_kernel<<<768, 256, 0, stream>>>(xq, xk, xv, Wq, bq, Wk, bk, Wv, bv,
                                      qbuf, kbuf, vTbuf);

  // Kernel 2: one block per (b,h,q-tile) = 2*16*128 = 4096 blocks
  // LDS: scores 128KB + 32 stat floats + pad + 8KB async stage
  const size_t shmem = (16 * SLEN + 32) * sizeof(float) + 8192;
  attn_kernel<<<4096, 128, shmem, stream>>>(qbuf, kbuf, vTbuf, ctxbuf, attn);

  // Kernel 3: (4096/32) * 16 = 2048 waves -> 256 blocks of 8 waves
  out_kernel<<<256, 256, 0, stream>>>(ctxbuf, Wo, bo, out);
}
